// DGMAtention_25031069401251
// MI455X (gfx1250) — matrix operand order repaired and checked by reading, not compiled, not run on hardware
//
#include <hip/hip_runtime.h>
#include <hip/hip_bf16.h>
#include <stdint.h>

// ---------------- problem constants ----------------
#define DIMC   768
#define Hh     12
#define HDc    64
#define Nn     196
#define GRID14 14
#define COVc   12
#define COVF   144
#define COVP   160      // 144 padded to multiple of 32 for WMMA K
#define PGc    7
#define TNv    2
#define VNc    3
#define Bb     48
#define BSc    16       // b_s = 48/3
#define NGc    12       // TN*TN*V group size
#define NSc    49       // PG*PG
#define MROWS  (Bb*Nn)  // 9408
#define SCALEF 0.125f
#define SSCALE 12.0f
#define EPSN   1e-5f

typedef __bf16 bf16_t;
typedef __attribute__((ext_vector_type(16))) __bf16 v16bf;
typedef __attribute__((ext_vector_type(8)))  __bf16 v8bf;
typedef __attribute__((ext_vector_type(8)))  float  v8f;

// ---------------- bf16 helpers (bit-level, RNE) ----------------
__device__ __forceinline__ bf16_t f2bf(float f) {
  union { float f; uint32_t u; } v; v.f = f;
  uint32_t r = v.u + 0x7FFFu + ((v.u >> 16) & 1u);
  union { uint16_t u; bf16_t b; } o; o.u = (uint16_t)(r >> 16);
  return o.b;
}
__device__ __forceinline__ float bf2f(bf16_t b) {
  union { uint16_t u; bf16_t b; } i; i.b = b;
  union { uint32_t u; float f; } o; o.u = ((uint32_t)i.u) << 16;
  return o.f;
}
__device__ __forceinline__ int imin(int a, int b) { return a < b ? a : b; }

// ---------------- register-blocked MBxNB wave WMMA tile ----------------
// Arow[mi]: A row for this lane, sub-tile mi (row-major over k).
// Brow[ni]: B column for this lane, pre-transposed so k is contiguous.
// Per K-step: MB+NB fragment loads feed MB*NB WMMAs.
template <int MB, int NB>
__device__ __forceinline__ void wmma_block(const bf16_t* const (&Arow)[MB],
                                           const bf16_t* const (&Brow)[NB],
                                           int Ksteps, int half, v8f (&acc)[MB][NB]) {
  for (int kb = 0; kb < Ksteps; ++kb) {
    int k0 = kb * 32;
    v16bf a[MB], b[NB];
#pragma unroll
    for (int mi = 0; mi < MB; ++mi) {
      // A frag: elems 0..15 -> k0+half*16+i (16 contiguous k)
      a[mi] = *(const v16bf*)(Arow[mi] + k0 + half * 16);
    }
#pragma unroll
    for (int ni = 0; ni < NB; ++ni)
      b[ni] = *(const v16bf*)(Brow[ni] + k0 + half * 16);  // 16 contiguous k
#pragma unroll
    for (int mi = 0; mi < MB; ++mi)
#pragma unroll
      for (int ni = 0; ni < NB; ++ni)
        acc[mi][ni] = __builtin_amdgcn_wmma_f32_16x16x32_bf16(
            false, a[mi], false, b[ni], (short)0, acc[mi][ni], false, false);
  }
}

// group index g -> (batch, pre-sort row)
__device__ __forceinline__ void map_group(int g, int s1, int s2, int bs,
                                          int& bb, int& rr) {
  int t1 = g / (TNv * VNc);
  int t2 = (g / VNc) % TNv;
  int v  = g % VNc;
  bb = bs * VNc + v;
  rr = ((t1 * PGc + s1) * TNv + t2) * PGc + s2;
}

// ---------------- k1: convert x to bf16, transpose weights to bf16 ----------------
__global__ void k_convert(const float* __restrict__ x,
                          const float* __restrict__ Wq, const float* __restrict__ Wk,
                          const float* __restrict__ Wv, const float* __restrict__ Wp,
                          bf16_t* __restrict__ XBF, bf16_t* __restrict__ WT) {
  size_t stride = (size_t)gridDim.x * blockDim.x;
  size_t t0 = (size_t)blockIdx.x * blockDim.x + threadIdx.x;
  for (size_t i = t0; i < (size_t)MROWS * DIMC; i += stride) XBF[i] = f2bf(x[i]);
  const float* Ws[4] = { Wq, Wk, Wv, Wp };
  for (size_t e = t0; e < (size_t)4 * DIMC * DIMC; e += stride) {
    int wsel = (int)(e / (DIMC * DIMC));
    int rem  = (int)(e % (DIMC * DIMC));
    int c = rem / DIMC, kk = rem % DIMC;          // WT[wsel][c][kk] = W[kk][c]
    WT[e] = f2bf(Ws[wsel][(size_t)kk * DIMC + c]);
  }
}

// ---------------- k2: QKV projections (WMMA, 64x32 wave tiles, no spill) ----------------
__global__ void __launch_bounds__(256, 1)
k_qkv_gemm(const bf16_t* __restrict__ XBF, const bf16_t* __restrict__ WT,
           float* __restrict__ Q, float* __restrict__ K, float* __restrict__ V,
           bf16_t* __restrict__ QBF, bf16_t* __restrict__ KBF) {
  const int MT = MROWS / 64, NT = DIMC / 32;           // 147, 24
  const int TOT = 3 * MT * NT;                          // 10584 waves
  int w = blockIdx.x * (blockDim.x >> 5) + (threadIdx.x >> 5);
  if (w >= TOT) return;
  int mat = w / (MT * NT), rem = w % (MT * NT);
  int mt = rem / NT, nt = rem % NT;
  int lane = threadIdx.x & 31, half = lane >> 4, l15 = lane & 15;
  const bf16_t* Arow[4];
  const bf16_t* Brow[2];
#pragma unroll
  for (int mi = 0; mi < 4; ++mi)
    Arow[mi] = XBF + (size_t)(mt * 64 + mi * 16 + l15) * DIMC;
#pragma unroll
  for (int ni = 0; ni < 2; ++ni)
    Brow[ni] = WT + (size_t)mat * DIMC * DIMC + (size_t)(nt * 32 + ni * 16 + l15) * DIMC;
  v8f acc[4][2] = {};
  wmma_block<4, 2>(Arow, Brow, DIMC / 32, half, acc);
  float*  dst = (mat == 0) ? Q : ((mat == 1) ? K : V);
  bf16_t* dbf = (mat == 0) ? QBF : ((mat == 1) ? KBF : nullptr);
#pragma unroll
  for (int ni = 0; ni < 2; ++ni) {
    int col = nt * 32 + ni * 16 + l15;
    int h = col >> 6, d = col & 63;
#pragma unroll
    for (int mi = 0; mi < 4; ++mi) {
#pragma unroll
      for (int r = 0; r < 8; ++r) {
        int m = mt * 64 + mi * 16 + half * 8 + r;
        int b = m / Nn, n = m % Nn;
        size_t o = (((size_t)b * Hh + h) * Nn + n) * HDc + d;
        float vv = acc[mi][ni][r];
        dst[o] = vv;
        if (dbf) dbf[o] = f2bf(vv);
      }
    }
  }
}

// ---------------- k3: pooled projection + covariance (padded bf16) + |.|^2 ----------------
__global__ void k_cov(const float* __restrict__ Q, const float* __restrict__ K,
                      const float* __restrict__ Wcp, const float* __restrict__ bcp,
                      bf16_t* __restrict__ CQ, bf16_t* __restrict__ CK,
                      float* __restrict__ SA, float* __restrict__ SB) {
  const int NR = Bb * Hh * Nn;
  int t = blockIdx.x * blockDim.x + threadIdx.x;
  if (t >= 2 * NR) return;
  int which = t / NR, r = t % NR;
  const float* row = (which ? K : Q) + (size_t)r * HDc;
  float p[COVc];
#pragma unroll
  for (int c = 0; c < COVc; ++c) {
    float s = bcp[c];
    for (int d = 0; d < HDc; ++d) s += row[d] * Wcp[d * COVc + c];
    p[c] = s;
  }
  float m = 0.f;
#pragma unroll
  for (int c = 0; c < COVc; ++c) m += p[c];
  m *= (1.0f / COVc);
  float xc[COVc];
#pragma unroll
  for (int c = 0; c < COVc; ++c) xc[c] = p[c] - m;
  bf16_t* out = (which ? CK : CQ) + (size_t)r * COVP;
  float s2 = 0.f;
  const float inv11 = 1.0f / (COVc - 1);
  for (int c = 0; c < COVc; ++c)
#pragma unroll
    for (int e = 0; e < COVc; ++e) {
      float cv = xc[c] * xc[e] * inv11;
      out[c * COVc + e] = f2bf(cv);
      s2 += cv * cv;
    }
  for (int i = COVF; i < COVP; ++i) out[i] = f2bf(0.0f);
  (which ? SB : SA)[r] = s2;
}

// ---------------- k4: attn_e raw = (q.k^T) * scale  (WMMA, 32x32 wave tiles) ----------------
__global__ void __launch_bounds__(256, 1)
k_attn_e(const bf16_t* __restrict__ QBF, const bf16_t* __restrict__ KBF,
         float* __restrict__ AE) {
  const int T = 7;                       // ceil(196/32)
  const int TOT = Bb * Hh * T * T;       // 28224 waves
  int w = blockIdx.x * (blockDim.x >> 5) + (threadIdx.x >> 5);
  if (w >= TOT) return;
  int bh = w / (T * T), rem = w % (T * T);
  int it = rem / T, jt = rem % T;
  int lane = threadIdx.x & 31, half = lane >> 4, l15 = lane & 15;
  const bf16_t* Arow[2];
  const bf16_t* Brow[2];
#pragma unroll
  for (int mi = 0; mi < 2; ++mi)
    Arow[mi] = QBF + ((size_t)bh * Nn + imin(it * 32 + mi * 16 + l15, Nn - 1)) * HDc;
#pragma unroll
  for (int ni = 0; ni < 2; ++ni)
    Brow[ni] = KBF + ((size_t)bh * Nn + imin(jt * 32 + ni * 16 + l15, Nn - 1)) * HDc;
  v8f acc[2][2] = {};
  wmma_block<2, 2>(Arow, Brow, HDc / 32, half, acc);
#pragma unroll
  for (int ni = 0; ni < 2; ++ni) {
    int j = jt * 32 + ni * 16 + l15;
    if (j < Nn) {
#pragma unroll
      for (int mi = 0; mi < 2; ++mi) {
#pragma unroll
        for (int r = 0; r < 8; ++r) {
          int i = it * 32 + mi * 16 + half * 8 + r;
          if (i < Nn) AE[((size_t)bh * Nn + i) * Nn + j] = acc[mi][ni][r] * SCALEF;
        }
      }
    }
  }
}

// ---------------- k5: attn_s raw = cdist(cov_q, cov_k) * 12  (WMMA, K=160) ----------------
__global__ void __launch_bounds__(256, 1)
k_attn_s(const bf16_t* __restrict__ CQ, const bf16_t* __restrict__ CK,
         const float* __restrict__ SA, const float* __restrict__ SB,
         float* __restrict__ AS) {
  const int T = 7;
  const int TOT = Bb * Hh * T * T;
  int w = blockIdx.x * (blockDim.x >> 5) + (threadIdx.x >> 5);
  if (w >= TOT) return;
  int bh = w / (T * T), rem = w % (T * T);
  int it = rem / T, jt = rem % T;
  int lane = threadIdx.x & 31, half = lane >> 4, l15 = lane & 15;
  const bf16_t* Arow[2];
  const bf16_t* Brow[2];
#pragma unroll
  for (int mi = 0; mi < 2; ++mi)
    Arow[mi] = CQ + ((size_t)bh * Nn + imin(it * 32 + mi * 16 + l15, Nn - 1)) * COVP;
#pragma unroll
  for (int ni = 0; ni < 2; ++ni)
    Brow[ni] = CK + ((size_t)bh * Nn + imin(jt * 32 + ni * 16 + l15, Nn - 1)) * COVP;
  v8f acc[2][2] = {};
  wmma_block<2, 2>(Arow, Brow, COVP / 32, half, acc);
#pragma unroll
  for (int ni = 0; ni < 2; ++ni) {
    int j = jt * 32 + ni * 16 + l15;
    if (j < Nn) {
      float sbj = SB[(size_t)bh * Nn + j];
#pragma unroll
      for (int mi = 0; mi < 2; ++mi) {
#pragma unroll
        for (int r = 0; r < 8; ++r) {
          int i = it * 32 + mi * 16 + half * 8 + r;
          if (i < Nn) {
            float d2 = SA[(size_t)bh * Nn + i] + sbj - 2.0f * acc[mi][ni][r];
            AS[((size_t)bh * Nn + i) * Nn + j] = sqrtf(fmaxf(d2, 1e-24f)) * SSCALE;
          }
        }
      }
    }
  }
}

// ---------------- k6: LN2d stats pass 1 (per (b,h,src) over NxN) ----------------
__global__ void k_stats1(const float* __restrict__ AE, const float* __restrict__ AS,
                         float* __restrict__ M1, float* __restrict__ IS1) {
  __shared__ float ss[256], sq[256];
  int slot = blockIdx.x, which = slot & 1, bh = slot >> 1;
  const float* src = (which ? AS : AE) + (size_t)bh * Nn * Nn;
  const int NE = Nn * Nn;
  float s = 0.f, q = 0.f;
  for (int e = threadIdx.x; e < NE; e += blockDim.x) { float v = src[e]; s += v; q += v * v; }
  ss[threadIdx.x] = s; sq[threadIdx.x] = q; __syncthreads();
  for (int st = 128; st > 0; st >>= 1) {
    if ((int)threadIdx.x < st) { ss[threadIdx.x] += ss[threadIdx.x + st]; sq[threadIdx.x] += sq[threadIdx.x + st]; }
    __syncthreads();
  }
  if (threadIdx.x == 0) {
    float m = ss[0] / NE;
    float var = fmaxf(sq[0] / NE - m * m, 0.0f);
    M1[slot] = m; IS1[slot] = rsqrtf(var + EPSN);
  }
}

// ---------------- k7: InstanceNorm stats of LN2d output (pass 2) ----------------
__global__ void k_stats2(const float* __restrict__ AE, const float* __restrict__ AS,
                         const float* __restrict__ lnw, const float* __restrict__ lnb,
                         const float* __restrict__ M1, const float* __restrict__ IS1,
                         float* __restrict__ M2, float* __restrict__ IS2) {
  __shared__ float ss[256], sq[256];
  int slot = blockIdx.x, which = slot & 1, bh = slot >> 1;
  const float* src = (which ? AS : AE) + (size_t)bh * Nn * Nn;
  float m1 = M1[slot], is1 = IS1[slot];
  const int NE = Nn * Nn;
  float s = 0.f, q = 0.f;
  for (int e = threadIdx.x; e < NE; e += blockDim.x) {
    float y = (src[e] - m1) * is1 * lnw[e] + lnb[e];
    s += y; q += y * y;
  }
  ss[threadIdx.x] = s; sq[threadIdx.x] = q; __syncthreads();
  for (int st = 128; st > 0; st >>= 1) {
    if ((int)threadIdx.x < st) { ss[threadIdx.x] += ss[threadIdx.x + st]; sq[threadIdx.x] += sq[threadIdx.x + st]; }
    __syncthreads();
  }
  if (threadIdx.x == 0) {
    float m = ss[0] / NE;
    float var = fmaxf(sq[0] / NE - m * m, 0.0f);
    M2[slot] = m; IS2[slot] = rsqrtf(var + EPSN);
  }
}

// ---------------- k8: diagonal of attn_group (only consumer of the mixed tensor) ----------------
__global__ void k_diag(const float* __restrict__ AE, const float* __restrict__ AS,
                       const float* __restrict__ lnw, const float* __restrict__ lnb,
                       const float* __restrict__ M1, const float* __restrict__ IS1,
                       const float* __restrict__ M2, const float* __restrict__ IS2,
                       const float* __restrict__ c2dw, const float* __restrict__ c2db,
                       float* __restrict__ DIAG) {
  int t = blockIdx.x * blockDim.x + threadIdx.x;
  if (t >= Bb * Nn) return;
  int i = t % Nn, b = t / Nn;
  float wii = lnw[i * Nn + i], bii = lnb[i * Nn + i];
  float eh[Hh], sh[Hh];
#pragma unroll
  for (int h = 0; h < Hh; ++h) {
    size_t bh = (size_t)b * Hh + h;
    int se = (int)bh * 2, sl = se + 1;
    float re = AE[(bh * Nn + i) * Nn + i];
    float rs = AS[(bh * Nn + i) * Nn + i];
    eh[h] = (((re - M1[se]) * IS1[se]) * wii + bii - M2[se]) * IS2[se];
    sh[h] = (((rs - M1[sl]) * IS1[sl]) * wii + bii - M2[sl]) * IS2[sl];
  }
#pragma unroll
  for (int o = 0; o < Hh; ++o) {
    float a = c2db[o];
#pragma unroll
    for (int h = 0; h < Hh; ++h)
      a += eh[h] * c2dw[o * 24 + h] + sh[h] * c2dw[o * 24 + 12 + h];
    DIAG[((size_t)b * Hh + o) * Nn + i] = a;
  }
}

// ---------------- k9: stable ascending argsort per (b,h) by rank counting ----------------
__global__ void k_argsort(const float* __restrict__ DIAG, int* __restrict__ IDX) {
  __shared__ float d[Nn];
  int bh = blockIdx.x, t = threadIdx.x;
  if (t < Nn) d[t] = DIAG[(size_t)bh * Nn + t];
  __syncthreads();
  if (t < Nn) {
    float di = d[t];
    int rank = 0;
    for (int j = 0; j < Nn; ++j) {
      float dj = d[j];
      rank += (dj < di) || (dj == di && j < t);
    }
    IDX[(size_t)bh * Nn + rank] = t;
  }
}

// ---------------- k10: grouped 12x12 dot products (gather on the fly) ----------------
__global__ void k_dots(const float* __restrict__ Q, const float* __restrict__ K,
                       const bf16_t* __restrict__ CQ, const bf16_t* __restrict__ CK,
                       const int* __restrict__ IDX,
                       float* __restrict__ DOTS, float* __restrict__ DCOV) {
  const int TOT = BSc * NSc * Hh * NGc * NGc;
  int t = blockIdx.x * blockDim.x + threadIdx.x;
  if (t >= TOT) return;
  int j = t % NGc; int r1 = t / NGc;
  int i = r1 % NGc; r1 /= NGc;
  int h = r1 % Hh;  r1 /= Hh;
  int s = r1 % NSc; int bs = r1 / NSc;
  int s1 = s / PGc, s2 = s % PGc;
  int bi, ri, bj, rj;
  map_group(i, s1, s2, bs, bi, ri);
  map_group(j, s1, s2, bs, bj, rj);
  int ni = IDX[((size_t)bi * Hh + h) * Nn + ri];
  int nj = IDX[((size_t)bj * Hh + h) * Nn + rj];
  const float* qr = Q + (((size_t)bi * Hh + h) * Nn + ni) * HDc;
  const float* kr = K + (((size_t)bj * Hh + h) * Nn + nj) * HDc;
  float acc = 0.f;
#pragma unroll 8
  for (int d = 0; d < HDc; ++d) acc += qr[d] * kr[d];
  DOTS[t] = acc * SCALEF;
  const bf16_t* cq = CQ + (((size_t)bi * Hh + h) * Nn + ri) * COVP;  // cov is NOT sorted
  const bf16_t* ck = CK + (((size_t)bj * Hh + h) * Nn + rj) * COVP;
  float ac = 0.f;
#pragma unroll 8
  for (int c = 0; c < COVF; ++c) ac += bf2f(cq[c]) * bf2f(ck[c]);
  DCOV[t] = ac * SSCALE;
}

// ---------------- k11: InstanceNorm3d stats per (b_s, channel of 98) over H*12*12 ----------------
__global__ void k_inorm3(const float* __restrict__ DOTS, const float* __restrict__ DCOV,
                         float* __restrict__ M3, float* __restrict__ IS3) {
  __shared__ float ss[128], sq[128];
  int blk = blockIdx.x;
  int c = blk % 98, bs = blk / 98;
  const int NE = Hh * NGc * NGc;   // 1728
  const float* base = (c < NSc)
      ? (DOTS + ((size_t)(bs * NSc + c) * Hh) * NGc * NGc)
      : (DCOV + ((size_t)(bs * NSc + (c - NSc)) * Hh) * NGc * NGc);
  float s = 0.f, q = 0.f;
  for (int e = threadIdx.x; e < NE; e += blockDim.x) { float v = base[e]; s += v; q += v * v; }
  ss[threadIdx.x] = s; sq[threadIdx.x] = q; __syncthreads();
  for (int st = 64; st > 0; st >>= 1) {
    if ((int)threadIdx.x < st) { ss[threadIdx.x] += ss[threadIdx.x + st]; sq[threadIdx.x] += sq[threadIdx.x + st]; }
    __syncthreads();
  }
  if (threadIdx.x == 0) {
    float m = ss[0] / NE;
    float var = fmaxf(sq[0] / NE - m * m, 0.0f);
    M3[blk] = m; IS3[blk] = rsqrtf(var + EPSN);
  }
}

// ---------------- k12: channel mix + softmax + attn @ v_cls ----------------
__global__ void k_gattn(const float* __restrict__ DOTS, const float* __restrict__ DCOV,
                        const float* __restrict__ M3, const float* __restrict__ IS3,
                        const float* __restrict__ c3dw, const float* __restrict__ c3db,
                        const float* __restrict__ V, const int* __restrict__ IDX,
                        float* __restrict__ OUTCLS) {
  const int TOT = BSc * NSc * Hh * NGc;   // 112896
  int t = blockIdx.x * blockDim.x + threadIdx.x;
  if (t >= TOT) return;
  int i = t % NGc; int r = t / NGc;
  int h = r % Hh;  r /= Hh;
  int o = r % NSc; int bs = r / NSc;
  float logit[NGc];
#pragma unroll
  for (int j = 0; j < NGc; ++j) logit[j] = c3db[o];
  for (int c = 0; c < NSc; ++c) {
    float w1 = c3dw[o * 98 + c];
    float m = M3[bs * 98 + c], is = IS3[bs * 98 + c];
    const float* src = DOTS + (((size_t)(bs * NSc + c) * Hh + h) * NGc + i) * NGc;
#pragma unroll
    for (int j = 0; j < NGc; ++j) logit[j] += (src[j] - m) * is * w1;
  }
  for (int c = 0; c < NSc; ++c) {
    float w2 = c3dw[o * 98 + NSc + c];
    float m = M3[bs * 98 + NSc + c], is = IS3[bs * 98 + NSc + c];
    const float* src = DCOV + (((size_t)(bs * NSc + c) * Hh + h) * NGc + i) * NGc;
#pragma unroll
    for (int j = 0; j < NGc; ++j) logit[j] += (src[j] - m) * is * w2;
  }
  float mx = logit[0];
#pragma unroll
  for (int j = 1; j < NGc; ++j) mx = fmaxf(mx, logit[j]);
  float p[NGc], sum = 0.f;
#pragma unroll
  for (int j = 0; j < NGc; ++j) { p[j] = expf(logit[j] - mx); sum += p[j]; }
  float inv = 1.0f / sum;
#pragma unroll
  for (int j = 0; j < NGc; ++j) p[j] *= inv;
  int s1 = o / PGc, s2 = o % PGc;
  const float* vr[NGc];
#pragma unroll
  for (int j = 0; j < NGc; ++j) {
    int bb, rr; map_group(j, s1, s2, bs, bb, rr);
    int n = IDX[((size_t)bb * Hh + h) * Nn + rr];
    vr[j] = V + (((size_t)bb * Hh + h) * Nn + n) * HDc;
  }
  float* dst = OUTCLS + (((size_t)(bs * NSc + o) * Hh + h) * NGc + i) * HDc;
  for (int d = 0; d < HDc; ++d) {
    float a = 0.f;
#pragma unroll
    for (int j = 0; j < NGc; ++j) a += p[j] * vr[j][d];
    dst[d] = a;
  }
}

// ---------------- k13: ungroup + depthwise 3x3 conv + add -> bf16 rows ----------------
__global__ void k_fuse(const float* __restrict__ OUTCLS, const float* __restrict__ V,
                       const float* __restrict__ ifsw, const float* __restrict__ ifsb,
                       bf16_t* __restrict__ PREBF) {
  const int TOT = Bb * Hh * Nn * HDc;
  int t = blockIdx.x * blockDim.x + threadIdx.x;
  if (t >= TOT) return;
  int d = t & 63; int r = t >> 6;
  int n = r % Nn; r /= Nn;
  int h = r % Hh; int b = r / Hh;
  // n = ((t1*PG+s1)*TN+t2)*PG+s2
  int s2 = n % PGc; int r2 = n / PGc;
  int t2 = r2 % TNv; r2 /= TNv;
  int s1 = r2 % PGc; int t1 = r2 / PGc;
  int bs = b / VNc, v = b % VNc;
  int g = (t1 * TNv + t2) * VNc + v;
  int s = s1 * PGc + s2;
  float oval = OUTCLS[(((size_t)(bs * NSc + s) * Hh + h) * NGc + g) * HDc + d];
  // depthwise 3x3 SAME conv over 14x14 grid of v
  int y = n / GRID14, x = n % GRID14;
  int ch = h * HDc + d;
  float a = ifsb[ch];
  const float* vbase = V + (((size_t)b * Hh + h) * Nn) * HDc + d;
#pragma unroll
  for (int dy = -1; dy <= 1; ++dy)
#pragma unroll
    for (int dx = -1; dx <= 1; ++dx) {
      int yy = y + dy, xx = x + dx;
      if (yy >= 0 && yy < GRID14 && xx >= 0 && xx < GRID14)
        a += vbase[(size_t)(yy * GRID14 + xx) * HDc] * ifsw[ch * 9 + (dy + 1) * 3 + (dx + 1)];
    }
  PREBF[((size_t)(b * Nn + n)) * DIMC + ch] = f2bf(oval + a);
}

// ---------------- k14: final projection (WMMA, 64x32 wave tiles) + bias ----------------
__global__ void __launch_bounds__(256, 1)
k_final(const bf16_t* __restrict__ PREBF, const bf16_t* __restrict__ WpT,
        const float* __restrict__ bproj, float* __restrict__ OUT) {
  const int MT = MROWS / 64, NT = DIMC / 32;   // 147, 24
  const int TOT = MT * NT;                      // 3528 waves
  int w = blockIdx.x * (blockDim.x >> 5) + (threadIdx.x >> 5);
  if (w >= TOT) return;
  int mt = w / NT, nt = w % NT;
  int lane = threadIdx.x & 31, half = lane >> 4, l15 = lane & 15;
  const bf16_t* Arow[4];
  const bf16_t* Brow[2];
#pragma unroll
  for (int mi = 0; mi < 4; ++mi)
    Arow[mi] = PREBF + (size_t)(mt * 64 + mi * 16 + l15) * DIMC;
#pragma unroll
  for (int ni = 0; ni < 2; ++ni)
    Brow[ni] = WpT + (size_t)(nt * 32 + ni * 16 + l15) * DIMC;
  v8f acc[4][2] = {};
  wmma_block<4, 2>(Arow, Brow, DIMC / 32, half, acc);
#pragma unroll
  for (int ni = 0; ni < 2; ++ni) {
    int col = nt * 32 + ni * 16 + l15;
    float bias = bproj[col];
#pragma unroll
    for (int mi = 0; mi < 4; ++mi) {
#pragma unroll
      for (int r = 0; r < 8; ++r) {
        int m = mt * 64 + mi * 16 + half * 8 + r;
        OUT[(size_t)m * DIMC + col] = acc[mi][ni][r] + bias;
      }
    }
  }
}

// ---------------- launch ----------------
extern "C" void kernel_launch(void* const* d_in, const int* in_sizes, int n_in,
                              void* d_out, int out_size, void* d_ws, size_t ws_size,
                              hipStream_t stream) {
  (void)in_sizes; (void)n_in; (void)out_size; (void)ws_size;
  const float* x     = (const float*)d_in[0];
  const float* Wq    = (const float*)d_in[1];
  const float* Wk    = (const float*)d_in[2];
  const float* Wv    = (const float*)d_in[3];
  const float* Wp    = (const float*)d_in[4];
  const float* bproj = (const float*)d_in[5];
  const float* ifsw  = (const float*)d_in[6];
  const float* ifsb  = (const float*)d_in[7];
  const float* Wcp   = (const float*)d_in[8];
  const float* bcp   = (const float*)d_in[9];
  const float* lnw   = (const float*)d_in[10];
  const float* lnb   = (const float*)d_in[11];
  const float* c2dw  = (const float*)d_in[12];
  const float* c2db  = (const float*)d_in[13];
  const float* c3dw  = (const float*)d_in[14];
  const float* c3db  = (const float*)d_in[15];
  float* OUT = (float*)d_out;

  // workspace carve-up (256B aligned)
  char* base = (char*)d_ws;
  size_t off = 0;
  auto alloc = [&](size_t bytes) -> void* {
    void* p = base + off;
    off = (off + bytes + 255) & ~(size_t)255;
    return p;
  };
  const size_t NQKV = (size_t)Bb * Hh * Nn * HDc;        // 7,225,344
  const size_t NROW = (size_t)Bb * Hh * Nn;              // 112,896
  bf16_t* XBF  = (bf16_t*)alloc((size_t)MROWS * DIMC * sizeof(bf16_t));
  bf16_t* WT   = (bf16_t*)alloc((size_t)4 * DIMC * DIMC * sizeof(bf16_t));
  float*  Q    = (float*) alloc(NQKV * sizeof(float));
  float*  K    = (float*) alloc(NQKV * sizeof(float));
  float*  V    = (float*) alloc(NQKV * sizeof(float));
  bf16_t* QBF  = (bf16_t*)alloc(NQKV * sizeof(bf16_t));
  bf16_t* KBF  = (bf16_t*)alloc(NQKV * sizeof(bf16_t));
  bf16_t* CQ   = (bf16_t*)alloc(NROW * COVP * sizeof(bf16_t));
  bf16_t* CK   = (bf16_t*)alloc(NROW * COVP * sizeof(bf16_t));
  float*  SA   = (float*) alloc(NROW * sizeof(float));
  float*  SB   = (float*) alloc(NROW * sizeof(float));
  float*  AE   = (float*) alloc(NROW * Nn * sizeof(float));
  float*  AS   = (float*) alloc(NROW * Nn * sizeof(float));
  float*  M1   = (float*) alloc(1152 * sizeof(float));
  float*  IS1  = (float*) alloc(1152 * sizeof(float));
  float*  M2   = (float*) alloc(1152 * sizeof(float));
  float*  IS2  = (float*) alloc(1152 * sizeof(float));
  float*  DIAG = (float*) alloc(NROW * sizeof(float));
  int*    IDX  = (int*)   alloc(NROW * sizeof(int));
  float*  DOTS = (float*) alloc((size_t)BSc * NSc * Hh * NGc * NGc * sizeof(float));
  float*  DCOV = (float*) alloc((size_t)BSc * NSc * Hh * NGc * NGc * sizeof(float));
  float*  M3   = (float*) alloc((size_t)BSc * 98 * sizeof(float));
  float*  IS3  = (float*) alloc((size_t)BSc * 98 * sizeof(float));
  float*  OCLS = (float*) alloc(NQKV * sizeof(float));
  bf16_t* PREB = (bf16_t*)alloc(NQKV * sizeof(bf16_t));

  // 1. convert / transpose
  k_convert<<<4096, 256, 0, stream>>>(x, Wq, Wk, Wv, Wp, XBF, WT);
  // 2. QKV GEMMs (WMMA, 64x32 per wave)
  {
    int waves = 3 * (MROWS / 64) * (DIMC / 32);
    k_qkv_gemm<<<(waves + 7) / 8, 256, 0, stream>>>(XBF, WT, Q, K, V, QBF, KBF);
  }
  // 3. pooled covariances
  k_cov<<<(2 * (int)NROW + 255) / 256, 256, 0, stream>>>(Q, K, Wcp, bcp, CQ, CK, SA, SB);
  // 4/5. raw score matrices (WMMA, 32x32 per wave)
  {
    int waves = Bb * Hh * 7 * 7;
    k_attn_e<<<(waves + 7) / 8, 256, 0, stream>>>(QBF, KBF, AE);
    k_attn_s<<<(waves + 7) / 8, 256, 0, stream>>>(CQ, CK, SA, SB, AS);
  }
  // 6/7. two-pass normalization stats
  k_stats1<<<1152, 256, 0, stream>>>(AE, AS, M1, IS1);
  k_stats2<<<1152, 256, 0, stream>>>(AE, AS, lnw, lnb, M1, IS1, M2, IS2);
  // 8. diagonal of attn_group
  k_diag<<<(Bb * Nn + 255) / 256, 256, 0, stream>>>(AE, AS, lnw, lnb, M1, IS1, M2, IS2,
                                                    c2dw, c2db, DIAG);
  // 9. argsort
  k_argsort<<<Bb * Hh, 224, 0, stream>>>(DIAG, IDX);
  // 10. grouped dots
  {
    int tot = BSc * NSc * Hh * NGc * NGc;
    k_dots<<<(tot + 255) / 256, 256, 0, stream>>>(Q, K, CQ, CK, IDX, DOTS, DCOV);
  }
  // 11. instance-norm-3d stats
  k_inorm3<<<BSc * 98, 128, 0, stream>>>(DOTS, DCOV, M3, IS3);
  // 12. mix + softmax + attn@v
  {
    int tot = BSc * NSc * Hh * NGc;
    k_gattn<<<(tot + 255) / 256, 256, 0, stream>>>(DOTS, DCOV, M3, IS3, c3dw, c3db, V, IDX, OCLS);
  }
  // 13. ungroup + depthwise conv + add
  {
    int tot = (int)NQKV;
    k_fuse<<<(tot + 255) / 256, 256, 0, stream>>>(OCLS, V, ifsw, ifsb, PREB);
  }
  // 14. final projection (WMMA, 64x32 per wave)
  {
    int waves = (MROWS / 64) * (DIMC / 32);
    k_final<<<(waves + 7) / 8, 256, 0, stream>>>(PREB, WT + (size_t)3 * DIMC * DIMC, bproj, OUT);
  }
}
